// RSSMCore_81681688035440
// MI455X (gfx1250) — matrix operand order repaired and checked by reading, not compiled, not run on hardware
//
#include <hip/hip_runtime.h>
#include <hip/hip_bf16.h>
#include <math.h>

typedef __attribute__((ext_vector_type(16))) __bf16 v16bf;
typedef __attribute__((ext_vector_type(8)))  __bf16 v8bf;
typedef __attribute__((ext_vector_type(8)))  float  v8f;

#define T_STEPS 64
#define BB      64      // batch
#define EE      1024    // embed dim
#define A_DIM   6       // action dim
#define DD      2048    // GRU hidden
#define HH      1000    // MLP hidden
#define HPAD    1008    // HH padded to tile multiple
#define ZDIM    1024    // S*R
#define G3      6144    // 3*D

// ---------------------------------------------------------------------------
// WMMA tile accumulation: one 16x16 C tile, looping K in steps of 32.
// A:  row-major [64, K] bf16 (K % 32 == 0, row stride = K)
// Wt: TRANSPOSED weights, [Npad][K] bf16, Npad % 16 == 0, zero rows beyond N.
// All loads unconditional -> inner loop is 4x global_load_b128 + v_wmma.
// ---------------------------------------------------------------------------
__device__ inline v8f wmma_tiles(const __bf16* __restrict__ A,
                                 const __bf16* __restrict__ Wt,
                                 int K, int arow, int wcol, int half, v8f acc) {
  const __bf16* ar = A + (size_t)arow * K;
  const __bf16* wb = Wt + (size_t)wcol * K;
  for (int k0 = 0; k0 < K; k0 += 32) {
    // A-fragment (16-bit A 16x32 layout): lane half h holds K = {h*8..h*8+7, 16+h*8..+7}
    v8bf alo = *(const v8bf*)(ar + k0 + half * 8);
    v8bf ahi = *(const v8bf*)(ar + k0 + 16 + half * 8);
    // B-fragment (32x16 layout): lane half h holds K = {h*8..h*8+7, 16+h*8..+7}, matching A
    v8bf blo = *(const v8bf*)(wb + k0 + half * 8);
    v8bf bhi = *(const v8bf*)(wb + k0 + 16 + half * 8);
    v16bf af, bfv;
#pragma unroll
    for (int j = 0; j < 8; ++j) {
      af[j] = alo[j];  af[j + 8] = ahi[j];
      bfv[j] = blo[j]; bfv[j + 8] = bhi[j];
    }
    acc = __builtin_amdgcn_wmma_f32_16x16x32_bf16(false, af, false, bfv,
                                                  (short)0, acc, false, false);
  }
  return acc;
}

// C[64, Ncols] = A1@W1 (+ A2@W2) + bias ; each wave owns one 16x16 tile.
__global__ __launch_bounds__(256) void gemm_dual_k(
    const __bf16* __restrict__ A1, const __bf16* __restrict__ W1, int K1,
    const __bf16* __restrict__ A2, const __bf16* __restrict__ W2, int K2,
    const float* __restrict__ bias, float* __restrict__ C,
    int Ncols, int ldc, int ntiles) {
  int wid  = (blockIdx.x * 256 + threadIdx.x) >> 5;
  int lane = threadIdx.x & 31;
  int total = 4 * ntiles;                       // M = 64 -> 4 row tiles
  if (wid >= total) return;                     // whole-wave exit: EXEC all-1s for WMMA
  int mt = wid / ntiles, nt = wid - mt * ntiles;
  int half = lane >> 4, mr = lane & 15;
  int arow = mt * 16 + mr;
  int wcol = nt * 16 + mr;
  v8f acc;
#pragma unroll
  for (int r = 0; r < 8; ++r) acc[r] = 0.0f;
  acc = wmma_tiles(A1, W1, K1, arow, wcol, half, acc);
  if (A2) acc = wmma_tiles(A2, W2, K2, arow, wcol, half, acc);
  if (wcol < Ncols) {                           // store guard only
    float bv = bias ? bias[wcol] : 0.0f;
#pragma unroll
    for (int r = 0; r < 8; ++r) {
      int row = mt * 16 + half * 8 + r;         // C layout: VGPR r -> M = 8*half + r
      C[(size_t)row * ldc + wcol] = acc[r] + bv;
    }
  }
}

// fp32 [K,N] weight -> bf16 transposed, K-padded and N-padded: [Npad][Kpad]
__global__ void cvt_w_k(const float* __restrict__ src, __bf16* __restrict__ dst,
                        int K, int N, int Kpad, int Npad) {
  long long i = (long long)blockIdx.x * 256 + threadIdx.x;
  long long total = (long long)Npad * Kpad;
  if (i >= total) return;
  int c = (int)(i / Kpad);
  int r = (int)(i - (long long)c * Kpad);
  dst[i] = (r < K && c < N) ? (__bf16)src[(size_t)r * N + c] : (__bf16)0.0f;
}

__global__ void init_state_k(const float* __restrict__ h0, const float* __restrict__ z0,
                             float* __restrict__ h, float* __restrict__ z) {
  int i = blockIdx.x * 256 + threadIdx.x;
  if (i < BB * DD) h[i] = h0[i];
  if (i < BB * ZDIM) z[i] = z0[i];
}

// Per-step prep: reset-mask h (in place) and z, emit bf16 copies; convert e, pad a (6->32).
__global__ void prep_k(int t, const unsigned char* __restrict__ resets,
                       const float* __restrict__ embeds, const float* __restrict__ actions,
                       float* __restrict__ h, const float* __restrict__ z,
                       __bf16* __restrict__ h_b, __bf16* __restrict__ z_b,
                       __bf16* __restrict__ e_b, __bf16* __restrict__ a_b) {
  int i = blockIdx.x * 256 + threadIdx.x;
  int job = blockIdx.y;
  if (job == 0) {
    if (i < BB * DD) {
      int b = i / DD;
      float m = resets[t * BB + b] ? 0.0f : 1.0f;
      float v = h[i] * m;
      h[i] = v;
      h_b[i] = (__bf16)v;
    }
  } else if (job == 1) {
    if (i < BB * ZDIM) {
      int b = i / ZDIM;
      float m = resets[t * BB + b] ? 0.0f : 1.0f;
      z_b[i] = (__bf16)(z[i] * m);
    }
  } else if (job == 2) {
    if (i < BB * EE) e_b[i] = (__bf16)embeds[(size_t)t * BB * EE + i];
  } else {
    if (i < BB * 32) {
      int b = i >> 5, c = i & 31;
      a_b[i] = (c < A_DIM) ? (__bf16)actions[(size_t)t * BB * A_DIM + b * A_DIM + c]
                           : (__bf16)0.0f;
    }
  }
}

// Row LayerNorm (biased var, eps=1e-3) + ELU, emit zero-padded bf16 row [ldout].
__global__ __launch_bounds__(256) void ln_elu_k(const float* __restrict__ in, int ldin, int n,
                                                const float* __restrict__ g,
                                                const float* __restrict__ bta,
                                                __bf16* __restrict__ out, int ldout) {
  __shared__ float s1[256], s2[256];
  int row = blockIdx.x, tid = threadIdx.x;
  const float* x = in + (size_t)row * ldin;
  float s = 0.0f, ss = 0.0f;
  for (int i = tid; i < n; i += 256) { float v = x[i]; s += v; ss += v * v; }
  s1[tid] = s; s2[tid] = ss;
  __syncthreads();
  for (int o = 128; o > 0; o >>= 1) {
    if (tid < o) { s1[tid] += s1[tid + o]; s2[tid] += s2[tid + o]; }
    __syncthreads();
  }
  float mean = s1[0] / n;
  float var  = s2[0] / n - mean * mean;
  float inv  = rsqrtf(var + 1e-3f);
  for (int i = tid; i < ldout; i += 256) {
    float v = 0.0f;
    if (i < n) {
      float y = (x[i] - mean) * inv * g[i] + bta[i];
      v = (y > 0.0f) ? y : (expf(y) - 1.0f);
    }
    out[(size_t)row * ldout + i] = (__bf16)v;
  }
}

// GRU gate math (PyTorch order r,z,n); writes new h (f32 + bf16), features, h_last.
__global__ void gru_k(int t, const float* __restrict__ gi, const float* __restrict__ gh,
                      float* __restrict__ h, __bf16* __restrict__ h_b,
                      float* __restrict__ feat, float* __restrict__ h_last) {
  int i = blockIdx.x * 256 + threadIdx.x;
  if (i >= BB * DD) return;
  int b = i / DD, d = i - b * DD;
  size_t gb = (size_t)b * G3 + d;
  float r = 1.0f / (1.0f + expf(-(gi[gb] + gh[gb])));
  float u = 1.0f / (1.0f + expf(-(gi[gb + DD] + gh[gb + DD])));
  float nn = tanhf(gi[gb + 2 * DD] + r * gh[gb + 2 * DD]);
  float hn = (1.0f - u) * nn + u * h[i];
  h[i] = hn;
  h_b[i] = (__bf16)hn;
  feat[(size_t)t * BB * 3072 + (size_t)b * 3072 + d] = hn;
  h_last[i] = hn;
}

// Straight-through categorical: forward value is exactly one-hot(argmax(logit+gumbel)).
__global__ __launch_bounds__(256) void sample_k(int t, const float* __restrict__ posts,
                                                float* __restrict__ samples,
                                                float* __restrict__ feat,
                                                float* __restrict__ z,
                                                float* __restrict__ z_last) {
  int wid  = (blockIdx.x * 256 + threadIdx.x) >> 5;
  int lane = threadIdx.x & 31;
  if (wid >= BB * 32) return;
  int b = wid >> 5, s = wid & 31;
  size_t off = (size_t)t * BB * ZDIM + (size_t)b * ZDIM + s * 32;
  float logit = posts[off + lane];
  unsigned x = ((unsigned)t * 1315423911u) ^ ((unsigned)b * 2654435761u) ^
               ((unsigned)s * 40503u) ^ ((unsigned)lane * 97u) ^ 0x9E3779B9u;
  x ^= x >> 17; x *= 0xed5ad4bbu; x ^= x >> 11; x *= 0xac4c1b51u; x ^= x >> 15;
  float u01 = (float)(x >> 8) * (1.0f / 16777216.0f) + 1e-7f;
  float key = logit - logf(-logf(u01));
  int best = lane; float bk = key;
  for (int o = 16; o > 0; o >>= 1) {
    float ok = __shfl_xor(bk, o, 32);
    int   oi = __shfl_xor(best, o, 32);
    if (ok > bk || (ok == bk && oi < best)) { bk = ok; best = oi; }
  }
  float oh = (lane == best) ? 1.0f : 0.0f;
  samples[off + lane] = oh;
  feat[(size_t)t * BB * 3072 + (size_t)b * 3072 + DD + s * 32 + lane] = oh;
  size_t zi = (size_t)b * ZDIM + s * 32 + lane;
  z[zi] = oh;
  z_last[zi] = oh;
}

// ---------------------------------------------------------------------------
extern "C" void kernel_launch(void* const* d_in, const int* in_sizes, int n_in,
                              void* d_out, int out_size, void* d_ws, size_t ws_size,
                              hipStream_t stream) {
  (void)in_sizes; (void)n_in; (void)out_size; (void)ws_size;
  const float* embeds  = (const float*)d_in[0];
  const float* actions = (const float*)d_in[1];
  const unsigned char* resets = (const unsigned char*)d_in[2];
  const float* h0 = (const float*)d_in[3];
  const float* z0 = (const float*)d_in[4];
  const float* zw = (const float*)d_in[5];
  const float* zb = (const float*)d_in[6];
  const float* aw = (const float*)d_in[7];
  const float* in_g = (const float*)d_in[8];
  const float* in_b = (const float*)d_in[9];
  const float* gru_wi = (const float*)d_in[10];
  const float* gru_wh = (const float*)d_in[11];
  const float* gru_bi = (const float*)d_in[12];
  const float* gru_bh = (const float*)d_in[13];
  const float* ph_w = (const float*)d_in[14];
  const float* ph_b = (const float*)d_in[15];
  const float* pe_w = (const float*)d_in[16];
  const float* po_g = (const float*)d_in[17];
  const float* po_b = (const float*)d_in[18];
  const float* pp_w = (const float*)d_in[19];
  const float* pp_b = (const float*)d_in[20];
  const float* prh_w = (const float*)d_in[21];
  const float* prh_b = (const float*)d_in[22];
  const float* pr_g = (const float*)d_in[23];
  const float* pr_b = (const float*)d_in[24];
  const float* prp_w = (const float*)d_in[25];
  const float* prp_b = (const float*)d_in[26];

  float* out     = (float*)d_out;
  float* priors  = out;
  float* posts   = priors  + (size_t)T_STEPS * BB * ZDIM;
  float* samples = posts   + (size_t)T_STEPS * BB * ZDIM;
  float* feats   = samples + (size_t)T_STEPS * BB * ZDIM;
  float* h_last  = feats   + (size_t)T_STEPS * BB * 3072;
  float* z_last  = h_last  + (size_t)BB * DD;

  // ---- workspace carve (bf16 weights ~54MB fit in 192MB L2) ----
  char* p = (char*)d_ws;
  auto alloc = [&](size_t bytes) -> char* {
    char* r = p; p += (bytes + 255) & ~((size_t)255); return r;
  };
  __bf16* zw_t  = (__bf16*)alloc((size_t)HPAD * ZDIM * 2); // [1008][1024]
  __bf16* aw_t  = (__bf16*)alloc((size_t)HPAD * 32 * 2);   // [1008][32]
  __bf16* wi_t  = (__bf16*)alloc((size_t)G3 * 1024 * 2);   // [6144][1024]
  __bf16* wh_t  = (__bf16*)alloc((size_t)G3 * DD * 2);     // [6144][2048]
  __bf16* ph_t  = (__bf16*)alloc((size_t)HPAD * DD * 2);   // [1008][2048]
  __bf16* pe_t  = (__bf16*)alloc((size_t)HPAD * EE * 2);   // [1008][1024]
  __bf16* pp_t  = (__bf16*)alloc((size_t)ZDIM * 1024 * 2); // [1024][1024]
  __bf16* prh_t = (__bf16*)alloc((size_t)HPAD * DD * 2);   // [1008][2048]
  __bf16* prp_t = (__bf16*)alloc((size_t)ZDIM * 1024 * 2); // [1024][1024]
  float*  h_s   = (float*)alloc((size_t)BB * DD * 4);
  float*  z_s   = (float*)alloc((size_t)BB * ZDIM * 4);
  __bf16* h_bb  = (__bf16*)alloc((size_t)BB * DD * 2);
  __bf16* z_bb  = (__bf16*)alloc((size_t)BB * ZDIM * 2);
  __bf16* e_bb  = (__bf16*)alloc((size_t)BB * EE * 2);
  __bf16* a_bb  = (__bf16*)alloc((size_t)BB * 32 * 2);
  __bf16* act_b = (__bf16*)alloc((size_t)BB * 1024 * 2);
  float*  tmp   = (float*)alloc((size_t)BB * HPAD * 4);
  float*  gi_s  = (float*)alloc((size_t)BB * G3 * 4);
  float*  gh_s  = (float*)alloc((size_t)BB * G3 * 4);

  auto cvt = [&](const float* src, __bf16* dst, int K, int N, int Kpad, int Npad) {
    long long tot = (long long)Npad * Kpad;
    hipLaunchKernelGGL(cvt_w_k, dim3((unsigned)((tot + 255) / 256)), dim3(256), 0, stream,
                       src, dst, K, N, Kpad, Npad);
  };
  cvt(zw,     zw_t,  ZDIM,  HH,   ZDIM, HPAD);
  cvt(aw,     aw_t,  A_DIM, HH,   32,   HPAD);
  cvt(gru_wi, wi_t,  HH,    G3,   1024, G3);
  cvt(gru_wh, wh_t,  DD,    G3,   DD,   G3);
  cvt(ph_w,   ph_t,  DD,    HH,   DD,   HPAD);
  cvt(pe_w,   pe_t,  EE,    HH,   EE,   HPAD);
  cvt(pp_w,   pp_t,  HH,    ZDIM, 1024, ZDIM);
  cvt(prh_w,  prh_t, DD,    HH,   DD,   HPAD);
  cvt(prp_w,  prp_t, HH,    ZDIM, 1024, ZDIM);

  hipLaunchKernelGGL(init_state_k, dim3((BB * DD + 255) / 256), dim3(256), 0, stream,
                     h0, z0, h_s, z_s);

  auto gemm = [&](const __bf16* A1, const __bf16* W1, int K1,
                  const __bf16* A2, const __bf16* W2, int K2,
                  const float* bias, float* C, int Ncols, int ldc) {
    int ntiles = (Ncols + 15) / 16;
    int blocks = (4 * ntiles + 7) / 8;          // 8 waves / 256-thread block
    hipLaunchKernelGGL(gemm_dual_k, dim3(blocks), dim3(256), 0, stream,
                       A1, W1, K1, A2, W2, K2, bias, C, Ncols, ldc, ntiles);
  };

  for (int t = 0; t < T_STEPS; ++t) {
    hipLaunchKernelGGL(prep_k, dim3((BB * DD + 255) / 256, 4), dim3(256), 0, stream,
                       t, resets, embeds, actions, h_s, z_s, h_bb, z_bb, e_bb, a_bb);
    // input stage: elu(ln(z@zw + a@aw + zb))
    gemm(z_bb, zw_t, ZDIM, a_bb, aw_t, 32, zb, tmp, HH, HPAD);
    hipLaunchKernelGGL(ln_elu_k, dim3(BB), dim3(256), 0, stream, tmp, HPAD, HH, in_g, in_b, act_b, 1024);
    // GRU gates
    gemm(act_b, wi_t, 1024, (const __bf16*)nullptr, (const __bf16*)nullptr, 0, gru_bi, gi_s, G3, G3);
    gemm(h_bb,  wh_t, DD,   (const __bf16*)nullptr, (const __bf16*)nullptr, 0, gru_bh, gh_s, G3, G3);
    hipLaunchKernelGGL(gru_k, dim3((BB * DD + 255) / 256), dim3(256), 0, stream,
                       t, gi_s, gh_s, h_s, h_bb, feats, h_last);
    // posterior head
    gemm(h_bb, ph_t, DD, e_bb, pe_t, EE, ph_b, tmp, HH, HPAD);
    hipLaunchKernelGGL(ln_elu_k, dim3(BB), dim3(256), 0, stream, tmp, HPAD, HH, po_g, po_b, act_b, 1024);
    gemm(act_b, pp_t, 1024, (const __bf16*)nullptr, (const __bf16*)nullptr, 0, pp_b,
         posts + (size_t)t * BB * ZDIM, ZDIM, ZDIM);
    hipLaunchKernelGGL(sample_k, dim3((BB * 32 * 32 + 255) / 256), dim3(256), 0, stream,
                       t, posts, samples, feats, z_s, z_last);
    // prior head (same as batched prior, computed per step to save workspace)
    gemm(h_bb, prh_t, DD, (const __bf16*)nullptr, (const __bf16*)nullptr, 0, prh_b, tmp, HH, HPAD);
    hipLaunchKernelGGL(ln_elu_k, dim3(BB), dim3(256), 0, stream, tmp, HPAD, HH, pr_g, pr_b, act_b, 1024);
    gemm(act_b, prp_t, 1024, (const __bf16*)nullptr, (const __bf16*)nullptr, 0, prp_b,
         priors + (size_t)t * BB * ZDIM, ZDIM, ZDIM);
  }
}
